// HybridDrugTargetNet_45921790329012
// MI455X (gfx1250) — matrix operand order repaired and checked by reading, not compiled, not run on hardware
//
#include <hip/hip_runtime.h>

// ---------------------------------------------------------------------------
// Types for CDNA5 WMMA (wave32): v_wmma_f32_16x16x32_bf16
// ---------------------------------------------------------------------------
typedef __attribute__((ext_vector_type(16))) __bf16 v16bf;
typedef __attribute__((ext_vector_type(8)))  __bf16 v8bf;
typedef __attribute__((ext_vector_type(8)))  float  v8f;

__device__ __forceinline__ __bf16 f2bf(float f) {
    union { float f; unsigned u; } c; c.f = f;
    unsigned r = c.u + 0x7FFFu + ((c.u >> 16) & 1u);   // round-to-nearest-even
    unsigned short h = (unsigned short)(r >> 16);
    return __builtin_bit_cast(__bf16, h);
}

#define N_NODES  100000
#define N_EDGES  1600000
#define N_GRAPHS 512
#define SEQ_LEN  1024
#define EMB_DIM  128
#define CONV_K   8
#define CONV_O   64
#define CONV_L   1017   // SEQ_LEN - CONV_K + 1
#define CONV_KK  (EMB_DIM * CONV_K)   // 1024

// ---------------------------------------------------------------------------
// Operand preparation (one-time per call, memory-bound)
// ---------------------------------------------------------------------------
// out[r, c] = (c < K) ? in[r*K + c] : 0   (row-major, K padded to Kp)
__global__ void convert_pad_rows(const float* __restrict__ in, __bf16* __restrict__ out,
                                 long total, int K, int Kp)
{
    long idx = (long)blockIdx.x * blockDim.x + threadIdx.x;
    if (idx >= total) return;
    int r = (int)(idx / Kp), c = (int)(idx - (long)r * Kp);
    out[idx] = f2bf((c < K) ? in[(long)r * K + c] : 0.0f);
}

// Bt[n, k] = (k < K) ? W[k*N + n] : 0   (weights -> transposed bf16 [N, Kp])
__global__ void transpose_convert(const float* __restrict__ W, __bf16* __restrict__ Bt,
                                  long total, int K, int N, int Kp)
{
    long idx = (long)blockIdx.x * blockDim.x + threadIdx.x;
    if (idx >= total) return;
    int n = (int)(idx / Kp), k = (int)(idx - (long)n * Kp);
    Bt[idx] = f2bf((k < K) ? W[(long)k * N + n] : 0.0f);
}

// ---------------------------------------------------------------------------
// Direct-fragment WMMA GEMM: C[M,N] = A[M,Kp] * Bt[N,Kp]^T (+bias, +relu)
// A, Bt bf16, Kp % 32 == 0, N % 32 == 0.  blockDim (32,4): each wave computes
// one 16x32 tile (A fragment reused for two B tiles).  Every fragment is
// contiguous, aligned 16B/32B runs in global memory: no LDS, no guards.
// ---------------------------------------------------------------------------
__global__ void wmma_gemm_bf16(const __bf16* __restrict__ A,
                               const __bf16* __restrict__ Bt,
                               const float* __restrict__ bias,
                               float* __restrict__ C,
                               int M, int N, int Kp, int relu)
{
    const int lane  = threadIdx.x;
    const int mTile = blockIdx.y * blockDim.y + threadIdx.y;
    const int nBase = blockIdx.x * 32;
    const int mBase = mTile * 16;
    const int half  = lane >> 4;
    const int l15   = lane & 15;

    int r = mBase + l15;
    const int arow = (r < M) ? r : (M - 1);          // branch-free M clamp
    const __bf16* ap  = A  + (long)arow * Kp + (half << 4);
    const __bf16* bp0 = Bt + (long)(nBase + l15) * Kp + (half << 4);
    const __bf16* bp1 = bp0 + (long)16 * Kp;

    v8f acc0 = {}, acc1 = {};
    for (int kb = 0; kb < Kp; kb += 32) {
        v16bf a = *(const v16bf*)(ap + kb);          // A[row, kb+16h .. +15]
        v16bf b0 = *(const v16bf*)(bp0 + kb);        // Bt[col,   kb+16h .. +15]
        v16bf b1 = *(const v16bf*)(bp1 + kb);        // Bt[col+16, ...]
        acc0 = __builtin_amdgcn_wmma_f32_16x16x32_bf16(
                   false, a, false, b0, (short)0, acc0, false, false);
        acc1 = __builtin_amdgcn_wmma_f32_16x16x32_bf16(
                   false, a, false, b1, (short)0, acc1, false, false);
    }

    const int col0 = nBase + l15;
    const int col1 = col0 + 16;
    float bv0 = bias ? bias[col0] : 0.0f;
    float bv1 = bias ? bias[col1] : 0.0f;
    #pragma unroll
    for (int v = 0; v < 8; ++v) {
        int row = mBase + v + (half << 3);
        if (row < M) {
            float x0 = acc0[v] + bv0, x1 = acc1[v] + bv1;
            if (relu) { x0 = fmaxf(x0, 0.0f); x1 = fmaxf(x1, 0.0f); }
            C[(long)row * N + col0] = x0;
            C[(long)row * N + col1] = x1;
        }
    }
}

// ---------------------------------------------------------------------------
// Implicit-im2col conv1d(k=8) WMMA + fused relu + global max pool.
// A (im2col gather) staged to fragment-ordered LDS; B fragments are direct
// contiguous 32B loads from pre-converted conv_wt[64][1024] bf16.
// Block (32,4): 4 waves = 64 positions x 16 channels. Grid (4, 16, 512).
// ---------------------------------------------------------------------------
__global__ void conv_wmma_maxpool(const __bf16* __restrict__ eseq,
                                  const __bf16* __restrict__ conv_wt,
                                  const float* __restrict__ conv_b,
                                  float* __restrict__ pfeat)
{
    __shared__ __align__(32) __bf16 Asmem[4][32][16];   // fragment order, 4 KB

    const int lane   = threadIdx.x;
    const int wv     = threadIdx.y;
    const int tid    = lane + (wv << 5);
    const int oBase  = blockIdx.x * 16;
    const int pBase0 = blockIdx.y * 64;
    const int g      = blockIdx.z;
    const int half   = lane >> 4;
    const int l15    = lane & 15;
    const bool pFull = (pBase0 + 64) <= CONV_L;     // false only for last group

    const __bf16* eg = eseq + (long)g * SEQ_LEN * EMB_DIM;
    const __bf16* bp = conv_wt + (long)(oBase + l15) * CONV_KK + (half << 4);

    v8f acc = {};
    for (int kb = 0; kb < CONV_KK; kb += 32) {
        // ---- stage A: im2col element (p, k) = eseq[g, p + (k&7), k>>3] ----
        if (pFull) {
            #pragma unroll
            for (int c = 0; c < 16; ++c) {
                int s = c * 128 + tid, row = s >> 5, k = s & 31;
                int kk = kb + k, i = kk >> 3, t = kk & 7;
                __bf16 v = eg[(long)(pBase0 + row + t) * EMB_DIM + i];
                Asmem[row >> 4][(row & 15) + ((k >> 4) << 4)]
                     [k & 15] = v;
            }
        } else {
            #pragma unroll
            for (int c = 0; c < 16; ++c) {
                int s = c * 128 + tid, row = s >> 5, k = s & 31;
                int p = pBase0 + row, kk = kb + k, i = kk >> 3, t = kk & 7;
                int pc = (p < CONV_L) ? p : 0;      // clamp, zero after load
                __bf16 v = eg[(long)(pc + t) * EMB_DIM + i];
                if (p >= CONV_L) v = f2bf(0.0f);
                Asmem[row >> 4][(row & 15) + ((k >> 4) << 4)]
                     [k & 15] = v;
            }
        }
        __syncthreads();

        v16bf a = *(const v16bf*)&Asmem[wv][lane][0];   // 2x ds_load_b128
        v16bf b = *(const v16bf*)(bp + kb);             // 2x global b128
        acc = __builtin_amdgcn_wmma_f32_16x16x32_bf16(
                  false, a, false, b, (short)0, acc, false, false);
        __syncthreads();
    }

    const int o  = oBase + l15;                     // always < 64
    float cb = conv_b[o];
    #pragma unroll
    for (int v = 0; v < 8; ++v) {
        int prow = pBase0 + wv * 16 + v + (half << 3);
        if (prow < CONV_L) {
            float val = fmaxf(acc[v] + cb, 0.0f);   // relu => >= 0
            // non-negative IEEE floats order as unsigned ints
            atomicMax((unsigned int*)&pfeat[g * CONV_O + o], __float_as_uint(val));
        }
    }
}

// ---------------------------------------------------------------------------
// GCN support kernels (memory-bound; atomics resolve in 192MB L2)
// ---------------------------------------------------------------------------
__global__ void deg_count_kernel(const int* __restrict__ dst, float* __restrict__ deg, int E)
{
    int e = blockIdx.x * blockDim.x + threadIdx.x;
    if (e < E) atomicAdd(&deg[dst[e]], 1.0f);
}

__global__ void dinv_kernel(const float* __restrict__ deg, float* __restrict__ dinv, int n)
{
    int i = blockIdx.x * blockDim.x + threadIdx.x;
    if (i < n) dinv[i] = rsqrtf(deg[i] + 1.0f);         // +1 self loop
}

__global__ void edge_norm_kernel(const int* __restrict__ src, const int* __restrict__ dst,
                                 const float* __restrict__ dinv, float* __restrict__ norm, int E)
{
    int e = blockIdx.x * blockDim.x + threadIdx.x;
    if (e < E) norm[e] = dinv[src[e]] * dinv[dst[e]];
}

// agg[dst] += xw[src] * norm  (consecutive threads -> consecutive features)
__global__ void edge_scatter_kernel(const int* __restrict__ src, const int* __restrict__ dst,
                                    const float* __restrict__ norm,
                                    const float* __restrict__ xw, float* __restrict__ agg,
                                    long total, int logF)
{
    long idx = (long)blockIdx.x * blockDim.x + threadIdx.x;
    if (idx >= total) return;
    int e = (int)(idx >> logF);
    int f = (int)(idx & ((1 << logF) - 1));
    int s = src[e], d = dst[e];
    atomicAdd(&agg[((long)d << logF) + f], xw[((long)s << logF) + f] * norm[e]);
}

// h = relu(agg + xw/(deg+1) + bias), in place over agg; optional bf16 copy
__global__ void gcn_finalize_kernel(float* __restrict__ agg, const float* __restrict__ xw,
                                    const float* __restrict__ deg, const float* __restrict__ bias,
                                    __bf16* __restrict__ hbf, long total, int logF)
{
    long idx = (long)blockIdx.x * blockDim.x + threadIdx.x;
    if (idx >= total) return;
    int f    = (int)(idx & ((1 << logF) - 1));
    int node = (int)(idx >> logF);
    float d  = deg[node] + 1.0f;
    float v  = fmaxf(agg[idx] + xw[idx] * (1.0f / d) + bias[f], 0.0f);
    agg[idx] = v;
    if (hbf) hbf[idx] = f2bf(v);
}

// segment mean pool (sum + count via atomics)
__global__ void pool_sum_kernel(const float* __restrict__ h, const int* __restrict__ batch,
                                float* __restrict__ sums, float* __restrict__ counts, long total)
{
    long idx = (long)blockIdx.x * blockDim.x + threadIdx.x;
    if (idx >= total) return;
    int f    = (int)(idx & 255);
    int node = (int)(idx >> 8);
    int g    = batch[node];
    atomicAdd(&sums[(long)g * 256 + f], h[idx]);
    if (f == 0) atomicAdd(&counts[g], 1.0f);
}

__global__ void embed_kernel(const int* __restrict__ seq, const float* __restrict__ emb,
                             __bf16* __restrict__ eseq, long total)
{
    long idx = (long)blockIdx.x * blockDim.x + threadIdx.x;
    if (idx >= total) return;
    int  i  = (int)(idx & (EMB_DIM - 1));
    long bl = idx >> 7;                                  // (b*SEQ_LEN + l)
    int tok = seq[bl];
    eseq[idx] = f2bf(emb[tok * EMB_DIM + i]);
}

// combined bf16 [512, 320]: [0:256) ligand mean, [256:320) protein max-pool
__global__ void combine_kernel(const float* __restrict__ sums, const float* __restrict__ counts,
                               const float* __restrict__ pfeat, __bf16* __restrict__ combined)
{
    int idx = blockIdx.x * blockDim.x + threadIdx.x;
    if (idx >= N_GRAPHS * 320) return;
    int g = idx / 320, j = idx - g * 320;
    float v = (j < 256) ? sums[(long)g * 256 + j] / fmaxf(counts[g], 1.0f)
                        : pfeat[g * CONV_O + (j - 256)];
    combined[idx] = f2bf(v);
}

__global__ void fc2_kernel(const float* __restrict__ hfc, const float* __restrict__ w,
                           const float* __restrict__ b, float* __restrict__ out)
{
    int g = blockIdx.x * blockDim.x + threadIdx.x;
    if (g >= N_GRAPHS) return;
    float s = 0.0f;
    #pragma unroll 8
    for (int k = 0; k < 512; ++k) s += hfc[(long)g * 512 + k] * w[k];
    out[g] = s + b[0];
}

// ---------------------------------------------------------------------------
// Host side
// ---------------------------------------------------------------------------
static inline size_t alignup(size_t x) { return (x + 255) & ~(size_t)255; }
static inline int    cdiv(long a, long b) { return (int)((a + b - 1) / b); }

extern "C" void kernel_launch(void* const* d_in, const int* in_sizes, int n_in,
                              void* d_out, int out_size, void* d_ws, size_t ws_size,
                              hipStream_t stream)
{
    const float* x       = (const float*)d_in[0];   // [100000, 78]
    const int*   eidx    = (const int*)  d_in[1];   // [2, E]
    const int*   batch   = (const int*)  d_in[2];   // [100000]
    const int*   pseq    = (const int*)  d_in[3];   // [512, 1024]
    const float* W1      = (const float*)d_in[4];   // [78, 128]
    const float* b1      = (const float*)d_in[5];
    const float* W2      = (const float*)d_in[6];   // [128, 256]
    const float* b2      = (const float*)d_in[7];
    const float* emb     = (const float*)d_in[8];   // [25, 128]
    const float* conv_w  = (const float*)d_in[9];   // [64, 128, 8]
    const float* conv_b  = (const float*)d_in[10];
    const float* fc1_w   = (const float*)d_in[11];  // [320, 512]
    const float* fc1_b   = (const float*)d_in[12];
    const float* fc2_w   = (const float*)d_in[13];  // [512, 1]
    const float* fc2_b   = (const float*)d_in[14];
    float*       out     = (float*)d_out;           // [512]

    const int E = in_sizes[1] / 2;
    const int* esrc = eidx;
    const int* edst = eidx + E;

    // ---- workspace layout: zero-init region first (single memset) ----
    char* base = (char*)d_ws;
    size_t off = 0;
    float* deg    = (float*)(base + off); off = alignup(off + (size_t)N_NODES * 4);
    float* agg1   = (float*)(base + off); off = alignup(off + (size_t)N_NODES * 128 * 4);
    float* agg2   = (float*)(base + off); off = alignup(off + (size_t)N_NODES * 256 * 4);
    float* sums   = (float*)(base + off); off = alignup(off + (size_t)N_GRAPHS * 256 * 4);
    float* counts = (float*)(base + off); off = alignup(off + (size_t)N_GRAPHS * 4);
    float* pfeat  = (float*)(base + off); off = alignup(off + (size_t)N_GRAPHS * CONV_O * 4);
    size_t zero_bytes = off;
    float*  dinv   = (float*)(base + off);  off = alignup(off + (size_t)N_NODES * 4);
    float*  enorm  = (float*)(base + off);  off = alignup(off + (size_t)E * 4);
    float*  xw1    = (float*)(base + off);  off = alignup(off + (size_t)N_NODES * 128 * 4);
    float*  xw2    = (float*)(base + off);  off = alignup(off + (size_t)N_NODES * 256 * 4);
    float*  hfc    = (float*)(base + off);  off = alignup(off + (size_t)N_GRAPHS * 512 * 4);
    __bf16* xpad   = (__bf16*)(base + off); off = alignup(off + (size_t)N_NODES * 96 * 2);
    __bf16* h1bf   = (__bf16*)(base + off); off = alignup(off + (size_t)N_NODES * 128 * 2);
    __bf16* eseq   = (__bf16*)(base + off); off = alignup(off + (size_t)N_GRAPHS * SEQ_LEN * EMB_DIM * 2);
    __bf16* W1t    = (__bf16*)(base + off); off = alignup(off + (size_t)128 * 96 * 2);
    __bf16* W2t    = (__bf16*)(base + off); off = alignup(off + (size_t)256 * 128 * 2);
    __bf16* fc1t   = (__bf16*)(base + off); off = alignup(off + (size_t)512 * 320 * 2);
    __bf16* convwt = (__bf16*)(base + off); off = alignup(off + (size_t)CONV_O * CONV_KK * 2);
    __bf16* comb   = (__bf16*)(base + off); off = alignup(off + (size_t)N_GRAPHS * 320 * 2);
    (void)ws_size;

    hipMemsetAsync(d_ws, 0, zero_bytes, stream);

    const int TB = 256;

    // ---- one-time operand prep: bf16 pad/transpose ----
    convert_pad_rows<<<cdiv((long)N_NODES * 96, TB), TB, 0, stream>>>(x, xpad,
                                                  (long)N_NODES * 96, 78, 96);
    transpose_convert<<<cdiv((long)128 * 96, TB), TB, 0, stream>>>(W1, W1t,
                                                  (long)128 * 96, 78, 128, 96);
    transpose_convert<<<cdiv((long)256 * 128, TB), TB, 0, stream>>>(W2, W2t,
                                                  (long)256 * 128, 128, 256, 128);
    transpose_convert<<<cdiv((long)512 * 320, TB), TB, 0, stream>>>(fc1_w, fc1t,
                                                  (long)512 * 320, 320, 512, 320);
    convert_pad_rows<<<cdiv((long)CONV_O * CONV_KK, TB), TB, 0, stream>>>(conv_w, convwt,
                                                  (long)CONV_O * CONV_KK, CONV_KK, CONV_KK);

    // ---- degrees + per-edge normalization coefficients ----
    deg_count_kernel<<<cdiv(E, TB), TB, 0, stream>>>(edst, deg, E);
    dinv_kernel<<<cdiv(N_NODES, TB), TB, 0, stream>>>(deg, dinv, N_NODES);
    edge_norm_kernel<<<cdiv(E, TB), TB, 0, stream>>>(esrc, edst, dinv, enorm, E);

    // ---- GCN layer 1: xw1 = x @ W1 ; scatter ; finalize ----
    {
        dim3 blk(32, 4);
        dim3 grd(128 / 32, cdiv(cdiv(N_NODES, 16), 4));
        wmma_gemm_bf16<<<grd, blk, 0, stream>>>(xpad, W1t, nullptr, xw1,
                                                N_NODES, 128, 96, 0);
    }
    {
        long total = (long)E * 128;
        edge_scatter_kernel<<<cdiv(total, TB), TB, 0, stream>>>(esrc, edst, enorm,
                                                                xw1, agg1, total, 7);
        long nt = (long)N_NODES * 128;
        gcn_finalize_kernel<<<cdiv(nt, TB), TB, 0, stream>>>(agg1, xw1, deg, b1,
                                                             h1bf, nt, 7);
    }

    // ---- GCN layer 2: xw2 = h1 @ W2 ; scatter ; finalize ----
    {
        dim3 blk(32, 4);
        dim3 grd(256 / 32, cdiv(cdiv(N_NODES, 16), 4));
        wmma_gemm_bf16<<<grd, blk, 0, stream>>>(h1bf, W2t, nullptr, xw2,
                                                N_NODES, 256, 128, 0);
    }
    {
        long total = (long)E * 256;
        edge_scatter_kernel<<<cdiv(total, TB), TB, 0, stream>>>(esrc, edst, enorm,
                                                                xw2, agg2, total, 8);
        long nt = (long)N_NODES * 256;
        gcn_finalize_kernel<<<cdiv(nt, TB), TB, 0, stream>>>(agg2, xw2, deg, b2,
                                                             nullptr, nt, 8);
    }

    // ---- graph mean pool ----
    {
        long total = (long)N_NODES * 256;
        pool_sum_kernel<<<cdiv(total, TB), TB, 0, stream>>>(agg2, batch, sums, counts, total);
    }

    // ---- protein branch: embed -> conv1d(WMMA, fused relu+maxpool) ----
    {
        long total = (long)N_GRAPHS * SEQ_LEN * EMB_DIM;
        embed_kernel<<<cdiv(total, TB), TB, 0, stream>>>(pseq, emb, eseq, total);
        dim3 grd(CONV_O / 16, 16, N_GRAPHS);        // 64 p-tiles / 4 waves
        conv_wmma_maxpool<<<grd, dim3(32, 4), 0, stream>>>(eseq, convwt, conv_b, pfeat);
    }

    // ---- concat -> fc1 (WMMA, fused bias+relu) -> fc2 ----
    combine_kernel<<<cdiv((long)N_GRAPHS * 320, TB), TB, 0, stream>>>(sums, counts, pfeat, comb);
    {
        dim3 blk(32, 4);
        dim3 grd(512 / 32, cdiv(cdiv(N_GRAPHS, 16), 4));
        wmma_gemm_bf16<<<grd, blk, 0, stream>>>(comb, fc1t, fc1_b, hfc,
                                                N_GRAPHS, 512, 320, 1);
    }
    fc2_kernel<<<cdiv(N_GRAPHS, TB), TB, 0, stream>>>(hfc, fc2_w, fc2_b, out);
    (void)out_size; (void)n_in;
}
